// CTI_87660282511587
// MI455X (gfx1250) — matrix operand order repaired and checked by reading, not compiled, not run on hardware
//
#include <hip/hip_runtime.h>

// ---------------------------------------------------------------------------
// CDNA5 (gfx1250) implementation of the rank-32 tri-attention + CTI reference.
// All heavy contractions run on v_wmma_f32_16x16x32_bf16 (wave32 WMMA).
// Round 2: double-buffered LDS pipelines, pair-packed (b32) LDS staging,
// and ASYNCcnt-tracked global_load_async_to_lds_b128 for the bf16 m-GEMM.
// ---------------------------------------------------------------------------

typedef __attribute__((ext_vector_type(16))) __bf16 v16bf;
typedef __attribute__((ext_vector_type(8)))  __bf16 v8bf;
typedef __attribute__((ext_vector_type(2)))  __bf16 v2bf;
typedef __attribute__((ext_vector_type(8)))  float  v8f;

__device__ __forceinline__ v8f wmma_bf16(v16bf a, v16bf b, v8f c) {
  // D = A(16x32) * B(32x16) + C, f32 accumulate
  return __builtin_amdgcn_wmma_f32_16x16x32_bf16(false, a, false, b, (short)0, c,
                                                 false, false);
}

// A fragment: 16x32 bf16 tile, row-major, leading dim `ld` elements.
// ISA 16-bit A layout: lanes 0-15 row M=lane, K in {0..7,16..23};
// lanes 16-31 same rows, K in {8..15,24..31}.
__device__ __forceinline__ v16bf frag_a(const __bf16* base, int ld) {
  const int lane = threadIdx.x & 31;
  const int row  = lane & 15;
  const int kh   = lane >> 4;
  const __bf16* p = base + row * ld + 8 * kh;
  v8bf x = *(const v8bf*)p;
  v8bf y = *(const v8bf*)(p + 16);
  v16bf r;
#pragma unroll
  for (int i = 0; i < 8; ++i) { r[i] = x[i]; r[i + 8] = y[i]; }
  return r;
}

// B fragment: 32x16 (KxN) tile stored N-major: layout [n][ld], K contiguous.
// Lanes 0-15: col N=lane, K in {0..7,16..23}; lanes 16-31: same cols, K in {8..15,24..31}.
__device__ __forceinline__ v16bf frag_b(const __bf16* base, int ld) {
  const int lane = threadIdx.x & 31;
  const int col  = lane & 15;
  const int kh   = lane >> 4;
  const __bf16* p = base + col * ld + 8 * kh;
  v8bf x = *(const v8bf*)p;
  v8bf y = *(const v8bf*)(p + 16);
  v16bf r;
#pragma unroll
  for (int i = 0; i < 8; ++i) { r[i] = x[i]; r[i + 8] = y[i]; }
  return r;
}

// Async VMEM -> LDS DMA (ASYNCcnt-tracked), per CDNA5 ISA.
// Generic LDS pointer low 32 bits == LDS byte offset (aperture rule, §10.2).
__device__ __forceinline__ void async_copy16(void* lds, const void* g) {
  asm volatile("global_load_async_to_lds_b128 %0, %1, off"
               :: "v"((unsigned)(unsigned long long)lds), "v"(g)
               : "memory");
}
__device__ __forceinline__ void wait_async0() {
  asm volatile("s_wait_asynccnt 0" ::: "memory");
}

__global__ void zero_f32(float* p, int n) {
  int i = blockIdx.x * blockDim.x + threadIdx.x;
  if (i < n) p[i] = 0.0f;
}

// ---------------------------------------------------------------------------
// Generic tiled GEMM: C(f32) = A(f32, MxK row-major) * B(f32, KxN).
// Block tile 128x128x32, 256 threads = 8 waves (4x2), wave tile 32x64.
// Double-buffered LDS; B staged as packed K-pairs (ds_store_b32).
// M,N multiples of 128; K multiple of 32 (true at all call sites).
// ---------------------------------------------------------------------------
__global__ __launch_bounds__(256) void gemm_f32_k(
    const float* __restrict__ A, const float* __restrict__ B, float* __restrict__ C,
    int K, int lda, int ldb, int ldc)
{
  constexpr int BM = 128, BN = 128, BK = 32;
  __shared__ __align__(16) __bf16 Asb[2][BM * BK];   // [m][k]
  __shared__ __align__(16) __bf16 Bsb[2][BN * BK];   // [n][k]
  const int tid  = threadIdx.x;
  const int wave = tid >> 5;
  const int lane = tid & 31;
  const int wm = wave >> 1;   // 0..3
  const int wn = wave & 1;    // 0..1
  A += (long long)blockIdx.x * BM * lda;
  C += (long long)blockIdx.x * BM * ldc + (long long)blockIdx.y * BN;
  const int n0 = blockIdx.y * BN;

  auto stageA = [&](int k0, int bufi) {
    const int row = tid >> 1, cb = (tid & 1) * 16;
    const float* g = A + (long long)row * lda + k0 + cb;
    __bf16* d = &Asb[bufi][row * BK + cb];
#pragma unroll
    for (int i = 0; i < 16; ++i) d[i] = (__bf16)g[i];
  };
  auto stageB = [&](int k0, int bufi) {
    const int kp = tid >> 4, nb = (tid & 15) * 8;     // k = 2kp, 2kp+1
    const float* g0 = B + (long long)(k0 + 2 * kp) * ldb + n0 + nb;
    const float* g1 = g0 + ldb;
#pragma unroll
    for (int i = 0; i < 8; ++i) {
      v2bf p; p[0] = (__bf16)g0[i]; p[1] = (__bf16)g1[i];
      *(v2bf*)&Bsb[bufi][(nb + i) * BK + 2 * kp] = p;
    }
  };

  v8f acc[2][4] = {};
  stageA(0, 0); stageB(0, 0);
  __syncthreads();
  const int NK = K / BK;
  for (int kt = 0; kt < NK; ++kt) {
    const int cur = kt & 1, nxt = cur ^ 1;
    if (kt + 1 < NK) { stageA((kt + 1) * BK, nxt); stageB((kt + 1) * BK, nxt); }
    v16bf af[2], bfr[4];
#pragma unroll
    for (int mi = 0; mi < 2; ++mi) af[mi]  = frag_a(&Asb[cur][(wm * 32 + mi * 16) * BK], BK);
#pragma unroll
    for (int ni = 0; ni < 4; ++ni) bfr[ni] = frag_b(&Bsb[cur][(wn * 64 + ni * 16) * BK], BK);
#pragma unroll
    for (int mi = 0; mi < 2; ++mi)
#pragma unroll
      for (int ni = 0; ni < 4; ++ni)
        acc[mi][ni] = wmma_bf16(af[mi], bfr[ni], acc[mi][ni]);
    __syncthreads();
  }

  const int col = lane & 15, kh = lane >> 4;
#pragma unroll
  for (int mi = 0; mi < 2; ++mi)
#pragma unroll
    for (int ni = 0; ni < 4; ++ni) {
      float* c = C + (long long)(wm * 32 + mi * 16) * ldc + wn * 64 + ni * 16 + col;
#pragma unroll
      for (int j = 0; j < 8; ++j)
        c[(long long)(j + 8 * kh) * ldc] = acc[mi][ni][j];
    }
}

// ---------------------------------------------------------------------------
// bf16 GEMM with async A staging: C(f32) = A(bf16) * B(bf16).
// A tiles DMA'd straight into LDS via global_load_async_to_lds_b128
// (no conversion needed); B tiles transposed+pair-packed by VALU.
// Used for m[b] = Vcat[b](256x2048) @ C2cat[b](2048x1024), per-batch strides.
// ---------------------------------------------------------------------------
__global__ __launch_bounds__(256) void gemm_bf16_async_k(
    const __bf16* __restrict__ A, const __bf16* __restrict__ B, float* __restrict__ C,
    int K, int lda, int ldb, int ldc,
    long long sA, long long sB, long long sC)
{
  constexpr int BM = 128, BN = 128, BK = 32;
  __shared__ __align__(16) __bf16 Asb[2][BM * BK];
  __shared__ __align__(16) __bf16 Bsb[2][BN * BK];   // [n][k]
  const int tid  = threadIdx.x;
  const int wave = tid >> 5;
  const int lane = tid & 31;
  const int wm = wave >> 1;
  const int wn = wave & 1;
  A += (long long)blockIdx.z * sA + (long long)blockIdx.x * BM * lda;
  B += (long long)blockIdx.z * sB;
  C += (long long)blockIdx.z * sC + (long long)blockIdx.x * BM * ldc
     + (long long)blockIdx.y * BN;
  const int n0 = blockIdx.y * BN;

  auto stageA_async = [&](int k0, int bufi) {
    // 128 rows x 64B; 512 x 16B chunks; 2 per thread
#pragma unroll
    for (int u = tid; u < 512; u += 256) {
      const int row = u >> 2, seg = (u & 3) * 8;   // 8 bf16 = 16B
      async_copy16(&Asb[bufi][row * BK + seg],
                   A + (long long)row * lda + k0 + seg);
    }
  };
  auto stageB = [&](int k0, int bufi) {
    const int kp = tid >> 4, nb = (tid & 15) * 8;
    const __bf16* g0 = B + (long long)(k0 + 2 * kp) * ldb + n0 + nb;
    const __bf16* g1 = g0 + ldb;
#pragma unroll
    for (int i = 0; i < 8; ++i) {
      v2bf p; p[0] = g0[i]; p[1] = g1[i];
      *(v2bf*)&Bsb[bufi][(nb + i) * BK + 2 * kp] = p;
    }
  };

  v8f acc[2][4] = {};
  stageA_async(0, 0); stageB(0, 0);
  wait_async0();
  __syncthreads();
  const int NK = K / BK;
  for (int kt = 0; kt < NK; ++kt) {
    const int cur = kt & 1, nxt = cur ^ 1;
    if (kt + 1 < NK) stageA_async((kt + 1) * BK, nxt);
    v16bf af[2], bfr[4];
#pragma unroll
    for (int mi = 0; mi < 2; ++mi) af[mi]  = frag_a(&Asb[cur][(wm * 32 + mi * 16) * BK], BK);
#pragma unroll
    for (int ni = 0; ni < 4; ++ni) bfr[ni] = frag_b(&Bsb[cur][(wn * 64 + ni * 16) * BK], BK);
#pragma unroll
    for (int mi = 0; mi < 2; ++mi)
#pragma unroll
      for (int ni = 0; ni < 4; ++ni)
        acc[mi][ni] = wmma_bf16(af[mi], bfr[ni], acc[mi][ni]);
    if (kt + 1 < NK) stageB((kt + 1) * BK, nxt);
    wait_async0();
    __syncthreads();
  }

  const int col = lane & 15, kh = lane >> 4;
#pragma unroll
  for (int mi = 0; mi < 2; ++mi)
#pragma unroll
    for (int ni = 0; ni < 4; ++ni) {
      float* c = C + (long long)(wm * 32 + mi * 16) * ldc + wn * 64 + ni * 16 + col;
#pragma unroll
      for (int j = 0; j < 8; ++j)
        c[(long long)(j + 8 * kh) * ldc] = acc[mi][ni][j];
    }
}

// ---------------------------------------------------------------------------
// v rank chain: vr_0 = V@wv0 (K=2048,N=64), then vr_r = vr_{r-1}@wv_rest[r-1].
// Each block owns 64 rows; all 32 ranks written to out[row][r*64+V] (bf16).
// 256 threads = 8 waves (4x2), wave tile 16x32.
// ---------------------------------------------------------------------------
__global__ __launch_bounds__(256) void chain_v_k(
    const float* __restrict__ V, const float* __restrict__ W0,
    const float* __restrict__ Wr, __bf16* __restrict__ out)
{
  __shared__ __align__(16) __bf16 As[64 * 32];   // [m][32]
  __shared__ __align__(16) __bf16 Bs[64 * 64];   // [n][k], ld varies per phase
  __shared__ __align__(16) __bf16 cur[64 * 64];  // current vr tile [m][64]
  const int tid  = threadIdx.x;
  const int wave = tid >> 5, lane = tid & 31;
  const int wm = wave >> 1, wn = wave & 1;
  const int col = lane & 15, kh = lane >> 4;
  const float* Ab = V + (long long)blockIdx.x * 64 * 2048;

  v8f acc[2] = {};
  // ---- phase 0: vr0 = Vblk(64x2048) @ W0(2048x64)
  for (int k0 = 0; k0 < 2048; k0 += 32) {
    { int row = tid >> 2, cb = (tid & 3) * 8;
      const float* g = Ab + (long long)row * 2048 + k0 + cb;
      __bf16* d = As + row * 32 + cb;
#pragma unroll
      for (int i = 0; i < 8; ++i) d[i] = (__bf16)g[i]; }
    { int kp = tid >> 5, nb = (tid & 31) * 2;        // k = 2kp,2kp+1; 2 n per thread
      const float* g0 = W0 + (long long)(k0 + 2 * kp) * 64 + nb;
      const float* g1 = g0 + 64;
#pragma unroll
      for (int i = 0; i < 2; ++i) {
        v2bf p; p[0] = (__bf16)g0[i]; p[1] = (__bf16)g1[i];
        *(v2bf*)&Bs[(nb + i) * 32 + 2 * kp] = p;
      } }
    __syncthreads();
    v16bf a = frag_a(As + (wm * 16) * 32, 32);
#pragma unroll
    for (int ni = 0; ni < 2; ++ni) {
      v16bf b = frag_b(Bs + (wn * 32 + ni * 16) * 32, 32);
      acc[ni] = wmma_bf16(a, b, acc[ni]);
    }
    __syncthreads();
  }
  // dump rank 0
#pragma unroll
  for (int ni = 0; ni < 2; ++ni)
#pragma unroll
    for (int j = 0; j < 8; ++j) {
      int m_ = wm * 16 + j + 8 * kh;
      int n_ = wn * 32 + ni * 16 + col;
      __bf16 v = (__bf16)acc[ni][j];
      cur[m_ * 64 + n_] = v;
      out[((long long)blockIdx.x * 64 + m_) * 2048 + n_] = v;
    }
  __syncthreads();

  // ---- chain ranks 1..31: new = cur(64x64) @ W(64x64)
  for (int r = 1; r < 32; ++r) {
    { int kp = tid >> 4, nb = (tid & 15) * 4;        // k = 2kp,2kp+1 (kp 0..15 -> K 0..31)
#pragma unroll
      for (int kk = 0; kk < 2; ++kk) {               // cover K 0..63 in two halves
        const float* g0 = Wr + (long long)(r - 1) * 4096
                        + (long long)(kk * 32 + 2 * kp) * 64 + nb;
        const float* g1 = g0 + 64;
#pragma unroll
        for (int i = 0; i < 4; ++i) {
          v2bf p; p[0] = (__bf16)g0[i]; p[1] = (__bf16)g1[i];
          *(v2bf*)&Bs[(nb + i) * 64 + kk * 32 + 2 * kp] = p;
        }
      } }
    __syncthreads();
    v8f na[2] = {};
#pragma unroll
    for (int k0 = 0; k0 < 64; k0 += 32) {
      v16bf a = frag_a(cur + (wm * 16) * 64 + k0, 64);
#pragma unroll
      for (int ni = 0; ni < 2; ++ni) {
        v16bf b = frag_b(Bs + (wn * 32 + ni * 16) * 64 + k0, 64);
        na[ni] = wmma_bf16(a, b, na[ni]);
      }
    }
    __syncthreads();   // all reads of cur complete before overwrite
#pragma unroll
    for (int ni = 0; ni < 2; ++ni)
#pragma unroll
      for (int j = 0; j < 8; ++j) {
        int m_ = wm * 16 + j + 8 * kh;
        int n_ = wn * 32 + ni * 16 + col;
        __bf16 v = (__bf16)na[ni][j];
        cur[m_ * 64 + n_] = v;
        out[((long long)blockIdx.x * 64 + m_) * 2048 + (long long)r * 64 + n_] = v;
      }
    __syncthreads();
  }
}

// ---------------------------------------------------------------------------
// q/g rank chain: one block per batch b (32 rows). out[b][r][row][col] bf16.
// 128 threads = 4 waves (2x2), wave tile 16x16.
// ---------------------------------------------------------------------------
__global__ __launch_bounds__(128) void chain_qg_k(
    const float* __restrict__ X, const float* __restrict__ W0,
    const float* __restrict__ Wr, __bf16* __restrict__ out)
{
  __shared__ __align__(16) __bf16 As[32 * 32];
  __shared__ __align__(16) __bf16 Bs[32 * 32];   // [n][k]
  __shared__ __align__(16) __bf16 cur[32 * 32];  // [m][32]
  const int tid  = threadIdx.x;
  const int wave = tid >> 5, lane = tid & 31;
  const int wm = wave >> 1, wn = wave & 1;
  const int col = lane & 15, kh = lane >> 4;
  const float* Xb = X + (long long)blockIdx.x * 32 * 1024;

  v8f acc = {};
  for (int k0 = 0; k0 < 1024; k0 += 32) {
    { int row = tid >> 2, cb = (tid & 3) * 8;
      const float* g = Xb + (long long)row * 1024 + k0 + cb;
      __bf16* d = As + row * 32 + cb;
#pragma unroll
      for (int i = 0; i < 8; ++i) d[i] = (__bf16)g[i]; }
    { int kp = tid >> 4, nb = (tid & 15) * 2;        // k = 2kp,2kp+1; 2 n each
      const float* g0 = W0 + (long long)(k0 + 2 * kp) * 32 + nb;
      const float* g1 = g0 + 32;
#pragma unroll
      for (int i = 0; i < 2; ++i) {
        v2bf p; p[0] = (__bf16)g0[i]; p[1] = (__bf16)g1[i];
        *(v2bf*)&Bs[(nb + i) * 32 + 2 * kp] = p;
      } }
    __syncthreads();
    v16bf a = frag_a(As + (wm * 16) * 32, 32);
    v16bf b = frag_b(Bs + (wn * 16) * 32, 32);
    acc = wmma_bf16(a, b, acc);
    __syncthreads();
  }
#pragma unroll
  for (int j = 0; j < 8; ++j) {
    int m_ = wm * 16 + j + 8 * kh;
    int n_ = wn * 16 + col;
    __bf16 v = (__bf16)acc[j];
    cur[m_ * 32 + n_] = v;
    out[((long long)blockIdx.x * 32) * 1024 + m_ * 32 + n_] = v;
  }
  __syncthreads();

  for (int r = 1; r < 32; ++r) {
    { int kp = tid >> 4, nb = (tid & 15) * 2;
      const float* g0 = Wr + (long long)(r - 1) * 1024 + (long long)(2 * kp) * 32 + nb;
      const float* g1 = g0 + 32;
#pragma unroll
      for (int i = 0; i < 2; ++i) {
        v2bf p; p[0] = (__bf16)g0[i]; p[1] = (__bf16)g1[i];
        *(v2bf*)&Bs[(nb + i) * 32 + 2 * kp] = p;
      } }
    __syncthreads();
    v16bf a = frag_a(cur + (wm * 16) * 32, 32);
    v16bf b = frag_b(Bs + (wn * 16) * 32, 32);
    v8f na = {};
    na = wmma_bf16(a, b, na);
    __syncthreads();
#pragma unroll
    for (int j = 0; j < 8; ++j) {
      int m_ = wm * 16 + j + 8 * kh;
      int n_ = wn * 16 + col;
      __bf16 v = (__bf16)na[j];
      cur[m_ * 32 + n_] = v;
      out[((long long)blockIdx.x * 32 + r) * 1024 + m_ * 32 + n_] = v;
    }
    __syncthreads();
  }
}

// ---------------------------------------------------------------------------
// C2cat[b][r*64+V][q*32+g] = sum_{Q,G} ga[r,V,Q,G]*qr[b,r,q,Q]*gr[b,r,g,G]
// block = (b, r, Vblk of 16); 4 waves, each handles 4 V values:
//   step1: C1[Q,g] = ga[V](QxG) @ gr^T      (B operand = gr[g][G] N-major)
//   step2: C2[q,g] = qr(qxQ)   @ C1         (C1 stored [g][Q] N-major)
// ---------------------------------------------------------------------------
__global__ __launch_bounds__(128) void build_c2_k(
    const float* __restrict__ GA, const __bf16* __restrict__ QR,
    const __bf16* __restrict__ GR, __bf16* __restrict__ C2)
{
  const int blk = blockIdx.x;          // b*128 + r*4 + vb
  const int vb  = blk & 3;
  const int r   = (blk >> 2) & 31;
  const int b   = blk >> 7;
  __shared__ __align__(16) __bf16 qs[32 * 32];        // qr[b][r]  [q][Q]
  __shared__ __align__(16) __bf16 gs[32 * 32];        // gr[b][r]  [g][G]
  __shared__ __align__(16) __bf16 gas[16 * 32 * 32];  // ga[r][Vblk..+16) [Q][G]
  __shared__ __align__(16) __bf16 c1s[16 * 32 * 32];  // per-V C1, [g][Q]
  const int tid  = threadIdx.x;
  const int wave = tid >> 5, lane = tid & 31;
  const int col = lane & 15, kh = lane >> 4;

  { int e = tid * 8;
    const __bf16* q = QR + ((long long)b * 32 + r) * 1024 + e;
    const __bf16* g = GR + ((long long)b * 32 + r) * 1024 + e;
#pragma unroll
    for (int i = 0; i < 8; ++i) { qs[e + i] = q[i]; gs[e + i] = g[i]; } }
  { const float* g = GA + ((long long)r * 64 + vb * 16) * 1024;
    for (int e = tid; e < 16 * 1024; e += 128) gas[e] = (__bf16)g[e]; }
  __syncthreads();

  for (int vv = 0; vv < 4; ++vv) {
    const int vloc = wave * 4 + vv;
    const __bf16* A1 = gas + vloc * 1024;   // [Q][G]
    __bf16* C1 = c1s + vloc * 1024;         // [g][Q]
    // step 1
    v8f s1[2][2] = {};
#pragma unroll
    for (int mi = 0; mi < 2; ++mi) {
      v16bf a = frag_a(A1 + mi * 16 * 32, 32);
#pragma unroll
      for (int ni = 0; ni < 2; ++ni) {
        v16bf bb = frag_b(gs + ni * 16 * 32, 32);
        s1[mi][ni] = wmma_bf16(a, bb, s1[mi][ni]);
      }
    }
    // store transposed as packed Q-pairs: C1[g][Q] (wave-local; LDS in-order)
#pragma unroll
    for (int mi = 0; mi < 2; ++mi)
#pragma unroll
      for (int ni = 0; ni < 2; ++ni)
#pragma unroll
        for (int j = 0; j < 8; j += 2) {
          v2bf p; p[0] = (__bf16)s1[mi][ni][j]; p[1] = (__bf16)s1[mi][ni][j + 1];
          *(v2bf*)&C1[(ni * 16 + col) * 32 + mi * 16 + 8 * kh + j] = p;
        }
    // step 2
    v8f s2[2][2] = {};
#pragma unroll
    for (int mi = 0; mi < 2; ++mi) {
      v16bf a = frag_a(qs + mi * 16 * 32, 32);
#pragma unroll
      for (int ni = 0; ni < 2; ++ni) {
        v16bf bb = frag_b(C1 + ni * 16 * 32, 32);
        s2[mi][ni] = wmma_bf16(a, bb, s2[mi][ni]);
      }
    }
    __bf16* orow = C2 + ((long long)b * 2048 + r * 64 + vb * 16 + vloc) * 1024;
#pragma unroll
    for (int mi = 0; mi < 2; ++mi)
#pragma unroll
      for (int ni = 0; ni < 2; ++ni)
#pragma unroll
        for (int j = 0; j < 8; ++j) {
          int q_ = mi * 16 + j + 8 * kh;
          int g_ = ni * 16 + col;
          orow[q_ * 32 + g_] = (__bf16)s2[mi][ni][j];
        }
  }
}

// ---------------------------------------------------------------------------
// Final: S[b] = m[b](256x1024) @ T2[b]^T with T2[qg][d]=zq[b,q,d]*zg[b,g,d]
// (B-tile built on the fly), epilogue: z[b,d] += sum_v S[v,d]*zv[b,v,d].
// Double-buffered; B staged as packed g-pairs.
// ---------------------------------------------------------------------------
__global__ __launch_bounds__(256) void final_k(
    const float* __restrict__ M_, const float* __restrict__ ZQ,
    const float* __restrict__ ZG, const float* __restrict__ ZV,
    float* __restrict__ Z)
{
  constexpr int BM = 128, BN = 128, BK = 32;
  __shared__ __align__(16) __bf16 Asb[2][BM * BK];
  __shared__ __align__(16) __bf16 Bsb[2][BN * BK];   // [n][k], k == g
  const int tid  = threadIdx.x;
  const int wave = tid >> 5, lane = tid & 31;
  const int wm = wave >> 1, wn = wave & 1;
  const int col = lane & 15, kh = lane >> 4;
  const int b  = blockIdx.z;
  const int n0 = blockIdx.y * BN;
  const float* A = M_ + ((long long)b * 256 + blockIdx.x * BM) * 1024;

  auto stageA = [&](int k0, int bufi) {
    const int row = tid >> 1, cb = (tid & 1) * 16;
    const float* g = A + (long long)row * 1024 + k0 + cb;
    __bf16* d = &Asb[bufi][row * BK + cb];
#pragma unroll
    for (int i = 0; i < 16; ++i) d[i] = (__bf16)g[i];
  };
  auto stageB = [&](int k0, int bufi) {
    const int q_ = k0 >> 5;                           // fixed q for this K-block
    const int gp = tid >> 4, nb = (tid & 15) * 8;     // g = 2gp, 2gp+1
    const float* zq  = ZQ + ((long long)b * 32 + q_) * 1024 + n0 + nb;
    const float* zg0 = ZG + ((long long)b * 32 + 2 * gp) * 1024 + n0 + nb;
    const float* zg1 = zg0 + 1024;
#pragma unroll
    for (int i = 0; i < 8; ++i) {
      v2bf p; p[0] = (__bf16)(zq[i] * zg0[i]); p[1] = (__bf16)(zq[i] * zg1[i]);
      *(v2bf*)&Bsb[bufi][(nb + i) * BK + 2 * gp] = p;
    }
  };

  v8f acc[2][4] = {};
  stageA(0, 0); stageB(0, 0);
  __syncthreads();
  for (int kt = 0; kt < 32; ++kt) {
    const int cur = kt & 1, nxt = cur ^ 1;
    if (kt + 1 < 32) { stageA((kt + 1) * BK, nxt); stageB((kt + 1) * BK, nxt); }
    v16bf af[2], bfr[4];
#pragma unroll
    for (int mi = 0; mi < 2; ++mi) af[mi]  = frag_a(&Asb[cur][(wm * 32 + mi * 16) * BK], BK);
#pragma unroll
    for (int ni = 0; ni < 4; ++ni) bfr[ni] = frag_b(&Bsb[cur][(wn * 64 + ni * 16) * BK], BK);
#pragma unroll
    for (int mi = 0; mi < 2; ++mi)
#pragma unroll
      for (int ni = 0; ni < 4; ++ni)
        acc[mi][ni] = wmma_bf16(af[mi], bfr[ni], acc[mi][ni]);
    __syncthreads();
  }

  // epilogue: multiply by zv, reduce over the 16 v-rows this lane owns, atomicAdd
#pragma unroll
  for (int ni = 0; ni < 4; ++ni) {
    const int d = n0 + wn * 64 + ni * 16 + col;
    float p = 0.0f;
#pragma unroll
    for (int mi = 0; mi < 2; ++mi)
#pragma unroll
      for (int j = 0; j < 8; ++j) {
        int vrow = blockIdx.x * BM + wm * 32 + mi * 16 + j + 8 * kh;
        p += acc[mi][ni][j] * ZV[((long long)b * 256 + vrow) * 1024 + d];
      }
    atomicAdd(&Z[b * 1024 + d], p);
  }
}

// ---------------------------------------------------------------------------
// Host launcher
// ---------------------------------------------------------------------------
extern "C" void kernel_launch(void* const* d_in, const int* in_sizes, int n_in,
                              void* d_out, int out_size, void* d_ws, size_t ws_size,
                              hipStream_t stream)
{
  (void)in_sizes; (void)n_in; (void)out_size; (void)ws_size;
  const float* v   = (const float*)d_in[0];
  const float* q   = (const float*)d_in[1];
  const float* g   = (const float*)d_in[2];
  const float* ga  = (const float*)d_in[3];
  const float* wv0 = (const float*)d_in[4];
  const float* wq0 = (const float*)d_in[5];
  const float* wg0 = (const float*)d_in[6];
  const float* wvr = (const float*)d_in[7];
  const float* wqr = (const float*)d_in[8];
  const float* wgr = (const float*)d_in[9];
  const float* wzv = (const float*)d_in[10];
  const float* wzq = (const float*)d_in[11];
  const float* wzg = (const float*)d_in[12];
  float* z = (float*)d_out;

  // workspace layout (bytes); total ~248 MB
  char* ws = (char*)d_ws;
  float*  zv = (float*) (ws);                     // [8192][1024] f32
  float*  zq = (float*) (ws +  33554432ull);      // [1024][1024] f32
  float*  zg = (float*) (ws +  37748736ull);      // [1024][1024] f32
  __bf16* vr = (__bf16*)(ws +  41943040ull);      // [8192][2048] bf16 (Vcat)
  __bf16* qr = (__bf16*)(ws +  75497472ull);      // [32][32][32][32] bf16
  __bf16* gr = (__bf16*)(ws +  77594624ull);      // [32][32][32][32] bf16
  __bf16* c2 = (__bf16*)(ws +  79691776ull);      // [32][2048][1024] bf16
  float*  m  = (float*) (ws + 213909504ull);      // [32][256][1024] f32

  // 0) z = 0 (d_out is poisoned; final_k accumulates with atomics)
  zero_f32<<<dim3(128), dim3(256), 0, stream>>>(z, 32 * 1024);

  // 1) zv = v @ wzv ; zq = q @ wzq ; zg = g @ wzg
  gemm_f32_k<<<dim3(64, 8, 1), dim3(256), 0, stream>>>(
      v, wzv, zv, 2048, 2048, 1024, 1024);
  gemm_f32_k<<<dim3(8, 8, 1), dim3(256), 0, stream>>>(
      q, wzq, zq, 1024, 1024, 1024, 1024);
  gemm_f32_k<<<dim3(8, 8, 1), dim3(256), 0, stream>>>(
      g, wzg, zg, 1024, 1024, 1024, 1024);

  // 2) rank-projection chains (all 32 ranks materialized, bf16)
  chain_v_k<<<dim3(128), dim3(256), 0, stream>>>(v, wv0, wvr, vr);
  chain_qg_k<<<dim3(32), dim3(128), 0, stream>>>(q, wq0, wqr, qr);
  chain_qg_k<<<dim3(32), dim3(128), 0, stream>>>(g, wg0, wgr, gr);

  // 3) C2cat[b][rV][qg] from ga, qr, gr
  build_c2_k<<<dim3(4096), dim3(128), 0, stream>>>(ga, qr, gr, c2);

  // 4) m[b] = Vcat[b] @ C2cat[b]  (async A staging, per-batch 256x1024x2048)
  gemm_bf16_async_k<<<dim3(2, 8, 32), dim3(256), 0, stream>>>(
      vr, c2, m, 2048, 2048, 1024, 1024,
      256LL * 2048, 2048LL * 1024, 256LL * 1024);

  // 5) z[b,d] = sum_v zv * (m[b] @ (zq*zg)^T)
  final_k<<<dim3(2, 8, 32), dim3(256), 0, stream>>>(m, zq, zg, zv, z);
}
